// RNN_83227876262378
// MI455X (gfx1250) — matrix operand order repaired and checked by reading, not compiled, not run on hardware
//
#include <hip/hip_runtime.h>
#include <hip/hip_bf16.h>

#define TT 2048
#define BB 512
#define HH 64
#define LSTR 80   // LDS row stride in halves: 160B -> 32B-aligned fragments, bank-staggered

typedef __attribute__((ext_vector_type(16))) _Float16 v16h;
typedef __attribute__((ext_vector_type(8)))  float    v8f;
typedef __attribute__((ext_vector_type(4)))  float    v4f;

__device__ __forceinline__ float fast_tanh(float x) {
#if defined(__has_builtin)
#if __has_builtin(__builtin_amdgcn_tanhf)
    return __builtin_amdgcn_tanhf(x);            // hw v_tanh_f32: 1 TRANS op
#else
    float e = __builtin_amdgcn_exp2f(x * 2.8853900817779268f); // 2*log2(e)
    return 1.0f - 2.0f * __builtin_amdgcn_rcpf(e + 1.0f);
#endif
#else
    float e = __builtin_amdgcn_exp2f(x * 2.8853900817779268f);
    return 1.0f - 2.0f * __builtin_amdgcn_rcpf(e + 1.0f);
#endif
}

// LDS-only split barrier: drain DS counter (our h stores/loads), then signal+wait
// the workgroup barrier. Avoids __syncthreads()'s global store/load counter drains,
// so wave 0's y global-store never stalls the whole workgroup.
__device__ __forceinline__ void lds_barrier() {
    asm volatile(
        "s_wait_dscnt 0x0\n\t"
        "s_barrier_signal -1\n\t"
        "s_barrier_wait -1"
        ::: "memory");
}

// Column -> LDS position permutation so each lane's A-fragment is contiguous.
// Row layout (halves): [blk0: c0-7,c16-23][blk1: c8-15,c24-31][blk2: c32-39,c48-55][blk3: c40-47,c56-63]
// Lane half=0 reads blk0 (a0) and blk2 (a1); half=1 reads blk1 and blk3 -> matches
// ISA 16-bit A 16x32 layout (lanes 0-15: K {0..7,16..23}; lanes 16-31: K {8..15,24..31}).
__device__ __forceinline__ int fpos(int n) {
    int block = ((n >> 5) << 1) | ((n >> 3) & 1);
    int pos   = (((n >> 4) & 1) << 3) | (n & 7);
    return block * 16 + pos;
}

__global__ __launch_bounds__(128)
void rnn_scan_kernel(const float* __restrict__ x,      // [T,B,1]
                     const float* __restrict__ h0,     // [1,B,H]
                     const float* __restrict__ W_ih,   // [H,1]
                     const float* __restrict__ W_hh,   // [H,H]
                     const float* __restrict__ b_ih,   // [H]
                     const float* __restrict__ b_hh,   // [H]
                     const float* __restrict__ W_lin,  // [1,H]
                     const float* __restrict__ b_lin,  // [1]
                     float* __restrict__ y,            // [T,B,1]
                     float* __restrict__ hlast)        // [1,B,H]
{
    __shared__ __attribute__((aligned(64))) _Float16 hbuf[2][16 * LSTR];

    const int tid  = threadIdx.x;
    const int w    = tid >> 5;        // wave id 0..3 -> hidden col tile
    const int lane = tid & 31;
    const int l16  = lane & 15;
    const int half = lane >> 4;       // 0|1
    const int coff = half * 8;        // A-fragment K sub-offset per ISA layout
    const int bbase = blockIdx.x * 16;

    // ---- per-lane loop constants ----
    const int   N    = w * 16 + l16;            // hidden column in C/D layout
    const float wih  = W_ih[N];
    const float bias = b_ih[N] + b_hh[N];
    const float blin = b_lin[0];
    const int   fN   = fpos(N);                 // LDS column slot for h-stores

    // B-fragments of W_hh^T (B[k][n] = W_hh[n][k]), ISA B layout:
    // lanes 0-15: K = Kbase+{0..7,16..23} ; lanes 16-31: K = Kbase+{8..15,24..31} ; N per lane
    v16h bf0, bf1;
    {
        const float* wr = W_hh + N * HH;
        const int k0 = half * 8;
        #pragma unroll
        for (int s = 0; s < 16; ++s) bf0[s] = (_Float16)wr[k0 + s + (s & 8)];
        #pragma unroll
        for (int s = 0; s < 16; ++s) bf1[s] = (_Float16)wr[32 + k0 + s + (s & 8)];
    }

    // W_lin arranged to match this lane's A-fragment slot order (wave 0 uses it)
    float wl[32];
    #pragma unroll
    for (int j = 0; j < 4; ++j)
        #pragma unroll
        for (int i = 0; i < 8; ++i)
            wl[j * 8 + i] = W_lin[j * 16 + coff + i];

    // ---- stage h0 (f32 -> f16, permuted columns) into hbuf[0] ----
    for (int i = tid; i < 16 * 64; i += 128) {
        int r = i >> 6, c = i & 63;
        hbuf[0][r * LSTR + fpos(c)] = (_Float16)h0[(bbase + r) * HH + c];
    }
    __syncthreads();

    // Per-lane loop-invariant LDS pointers (all in-loop addresses become
    // base + immediate-offset, no per-iteration address VALU).
    const int myrow = l16;
    const _Float16* rd0 = &hbuf[0][myrow * LSTR + half * 16];
    const _Float16* rd1 = &hbuf[1][myrow * LSTR + half * 16];
    _Float16*       st0 = &hbuf[0][(half * 8) * LSTR + fN];
    _Float16*       st1 = &hbuf[1][(half * 8) * LSTR + fN];

    // x slice this lane needs each step: rows bbase+coff .. +7
    const float* xr = x + bbase + coff;
    v4f xc0 = *(const v4f*)(xr);
    v4f xc1 = *(const v4f*)(xr + 4);
    const float* xq = xr + BB;        // prefetch pointer (t+1)

    float cw[8];                       // final h kept in f32 for epilogue

    auto step = [&](const _Float16* rd, _Float16* st, int t) {
        // A fragments: contiguous 32B loads straight into WMMA source regs
        v16h a0 = *(const v16h*)(rd);
        v16h a1 = *(const v16h*)(rd + 32);

        // C init: xp = x*W_ih + b_ih + b_hh
        v8f c;
        #pragma unroll
        for (int r = 0; r < 8; ++r) {
            float xv = (r < 4) ? xc0[r] : xc1[r - 4];
            c[r] = xv * wih + bias;
        }

        // prefetch next timestep's x off the critical path
        v4f xn0 = xc0, xn1 = xc1;
        if (t + 1 < TT) {
            xn0 = *(const v4f*)(xq);
            xn1 = *(const v4f*)(xq + 4);
        }

        // h_t @ W_hh^T : two chained WMMAs (K = 64)
        c = __builtin_amdgcn_wmma_f32_16x16x32_f16(false, a0, false, bf0, (short)0, c, false, false);
        c = __builtin_amdgcn_wmma_f32_16x16x32_f16(false, a1, false, bf1, (short)0, c, false, false);

        // tanh, write h_{t+1} (f16, permuted column slot) to the other buffer
        #pragma unroll
        for (int r = 0; r < 8; ++r) {
            float hv = fast_tanh(c[r]);
            cw[r] = hv;
            st[r * LSTR] = (_Float16)hv;
        }

        // y[t-1] = h_t . W_lin (wave 0 only; reuses its loaded A row)
        if (w == 0 && t > 0) {
            float s = 0.f;
            #pragma unroll
            for (int i = 0; i < 16; ++i)
                s += (float)a0[i] * wl[i] + (float)a1[i] * wl[16 + i];
            s += __shfl_xor(s, 16);
            if (half == 0) y[(size_t)(t - 1) * BB + bbase + l16] = s + blin;
        }

        lds_barrier();
        xc0 = xn0; xc1 = xn1;
        xq += BB;
    };

    for (int t = 0; t < TT; t += 2) {
        step(rd0, st1, t);        // read hbuf[0], write hbuf[1]
        step(rd1, st0, t + 1);    // read hbuf[1], write hbuf[0]
    }

    // ---- epilogue: h_last (f32 from registers) and y[T-1]; final h is in hbuf[0] ----
    #pragma unroll
    for (int r = 0; r < 8; ++r)
        hlast[(size_t)(bbase + r + half * 8) * HH + N] = cw[r];

    if (w == 0) {
        v16h a0 = *(const v16h*)(rd0);
        v16h a1 = *(const v16h*)(rd0 + 32);
        float s = 0.f;
        #pragma unroll
        for (int i = 0; i < 16; ++i)
            s += (float)a0[i] * wl[i] + (float)a1[i] * wl[16 + i];
        s += __shfl_xor(s, 16);
        if (half == 0) y[(size_t)(TT - 1) * BB + bbase + l16] = s + blin;
    }
}

extern "C" void kernel_launch(void* const* d_in, const int* in_sizes, int n_in,
                              void* d_out, int out_size, void* d_ws, size_t ws_size,
                              hipStream_t stream) {
    const float* x     = (const float*)d_in[0];
    const float* h0    = (const float*)d_in[1];
    const float* W_ih  = (const float*)d_in[2];
    const float* W_hh  = (const float*)d_in[3];
    const float* b_ih  = (const float*)d_in[4];
    const float* b_hh  = (const float*)d_in[5];
    const float* W_lin = (const float*)d_in[6];
    const float* b_lin = (const float*)d_in[7];

    float* y     = (float*)d_out;            // [T,B,1]
    float* hlast = y + (size_t)TT * BB;      // [1,B,H]

    dim3 grid(BB / 16);    // 32 workgroups, one per 16 batch rows
    dim3 block(128);       // 4 waves: one 16x16 output tile each
    rnn_scan_kernel<<<grid, block, 0, stream>>>(x, h0, W_ih, W_hh, b_ih, b_hh,
                                                W_lin, b_lin, y, hlast);
}
